// LSTM_SRU_64871186039273
// MI455X (gfx1250) — matrix operand order repaired and checked by reading, not compiled, not run on hardware
//
#include <hip/hip_runtime.h>
#include <hip/hip_bf16.h>

// Problem constants (T,B,I,H,L) = (512, 64, 1024, 1024, 2)
#define TSTEPS 512
#define BB 64
#define II 1024
#define HH 1024
#define KIN 1024
#define KTOT 2048   // KIN + HH

typedef __attribute__((ext_vector_type(16))) __bf16 v16bf;
typedef __attribute__((ext_vector_type(8)))  float  v8f;
typedef __attribute__((ext_vector_type(4)))  unsigned int u32x4;
typedef __attribute__((ext_vector_type(8)))  unsigned int u32x8;
typedef __attribute__((ext_vector_type(4)))  float  f32x4;

union AB32 { u32x4 q[2]; v16bf v; };   // 32-byte WMMA A/B fragment

__device__ __forceinline__ float sigmoidf_(float x) {
    return 1.0f / (1.0f + __expf(-x));
}

// fp32 -> bf16 conversion, 4 elements/thread (one-time cost; bf16 weights +
// activations then live in the 192MB L2 / stream at half the bytes)
__global__ void cvt_bf16x4_kernel(const float* __restrict__ src,
                                  __hip_bfloat16* __restrict__ dst, int n4) {
    int i = blockIdx.x * 256 + threadIdx.x;
    if (i < n4) {
        f32x4 v = reinterpret_cast<const f32x4*>(src)[i];
        union { __hip_bfloat16 h[4]; unsigned long long u; } p;
        p.h[0] = __float2bfloat16(v.x);
        p.h[1] = __float2bfloat16(v.y);
        p.h[2] = __float2bfloat16(v.z);
        p.h[3] = __float2bfloat16(v.w);
        *reinterpret_cast<unsigned long long*>(&dst[4 * i]) = p.u;
    }
}

// ---- Tensor Data Mover: stage a 64-row x 256-col bf16 tile (row stride 1024
// elements) from global into LDS. D# per CDNA5 ISA ch.8; issued once per wave,
// tracked by TENSORcnt. VADDR2/3 passed as zeroed groups (tile_dim2=0 -> 2D).
__device__ __forceinline__ void tdm_stage_64x256(unsigned lds_off,
                                                 const __hip_bfloat16* gsrc) {
    unsigned long long ga = (unsigned long long)(size_t)gsrc;
    u32x4 g0;
    g0.x = 1u;                                        // count=1 (valid), user desc
    g0.y = lds_off;                                   // lds_addr[31:0]
    g0.z = (unsigned)(ga & 0xffffffffu);              // global_addr[31:0]
    g0.w = (unsigned)((ga >> 32) & 0x01ffffffu)       // global_addr[56:32]
         | (2u << 30);                                // type=2 ("image")
    u32x8 g1;
    g1[0] = (1u << 16);           // workgroup_mask=0, data_size=1 (2 bytes)
    g1[1] = (1024u << 16);        // tensor_dim0 = 1024 (bits 79:48, lo16 here)
    g1[2] = (64u << 16);          // tensor_dim0 hi16=0 | tensor_dim1=64 lo16
    g1[3] = (256u << 16);         // tensor_dim1 hi16=0 | tile_dim0=256
    g1[4] = 64u;                  // tile_dim1=64, tile_dim2=0 (2D)
    g1[5] = 1024u;                // tensor_dim0_stride lo32 = 1024 elements
    g1[6] = 0u;                   // stride0 hi16 | tensor_dim1_stride lo16
    g1[7] = 0u;                   // tensor_dim1_stride (unused, 2D tile)
    u32x4 gz; gz.x = 0u; gz.y = 0u; gz.z = 0u; gz.w = 0u;
    asm volatile("tensor_load_to_lds %0, %1, %2, %3"
                 :: "s"(g0), "s"(g1), "s"(gz), "s"(gz)
                 : "memory");
}

// One K-slab (256 cols) of WMMA work: 8 x K=32 steps, 2 tiles/wave, 4 gate
// fragments (+ tx fragment when TX). All B fragments loaded before the WMMAs
// so the scheduler can clause the global_load_b128s and hide L2 latency.
template <bool TX>
__device__ __forceinline__ void slab_mma(const __hip_bfloat16* __restrict__ sA,
                                         v8f (&acc)[2][5],
                                         const __hip_bfloat16* __restrict__ Wall,
                                         const __hip_bfloat16* __restrict__ Wt,
                                         int kg0, int n0, int wave, int mloc,
                                         int khalf) {
    for (int kc = 0; kc < 8; ++kc) {
        const int kk = kc * 32;
#pragma unroll
        for (int ti = 0; ti < 2; ++ti) {
            const int tile = wave + ti * 8;      // 0..15
            const int mt = tile & 3;
            const int nt = tile >> 2;

            // A fragment (16x32 bf16) from LDS: half 0 -> K kk+[0..15], half 1 -> kk+[16..31]
            AB32 afr;
            const __hip_bfloat16* ap = &sA[(mt * 16 + mloc) * 256 + kk + khalf * 16];
            afr.q[0] = *reinterpret_cast<const u32x4*>(ap);
            afr.q[1] = *reinterpret_cast<const u32x4*>(ap + 8);

            // B fragments (32x16): lane&15 = N, lane>>4 selects K 16-group;
            // K is contiguous in the weight row -> one 32B read per lane.
            const int ncol = n0 + nt * 16 + mloc;
            const int kb   = kg0 + kk + khalf * 16;
            AB32 bfr[5];
#pragma unroll
            for (int f = 0; f < 4; ++f) {
                const __hip_bfloat16* bp = Wall + (size_t)(f * HH + ncol) * KTOT + kb;
                bfr[f].q[0] = *reinterpret_cast<const u32x4*>(bp);
                bfr[f].q[1] = *reinterpret_cast<const u32x4*>(bp + 8);
            }
            if (TX) {
                const __hip_bfloat16* bp = Wt + (size_t)ncol * KIN + kb;
                bfr[4].q[0] = *reinterpret_cast<const u32x4*>(bp);
                bfr[4].q[1] = *reinterpret_cast<const u32x4*>(bp + 8);
            }
#pragma unroll
            for (int f = 0; f < 4; ++f)
                acc[ti][f] = __builtin_amdgcn_wmma_f32_16x16x32_bf16(
                    false, afr.v, false, bfr[f].v, (short)0, acc[ti][f], false, false);
            if (TX)
                acc[ti][4] = __builtin_amdgcn_wmma_f32_16x16x32_bf16(
                    false, afr.v, false, bfr[4].v, (short)0, acc[ti][4], false, false);
        }
    }
}

// Fused LSTM cell step for one layer:
//   gates(i,f,o,g) = [inp | h_prev] @ W_all^T + b_all    (64 x 4096, K=2048)
//   tx             =  inp @ W_t^T + b_t                  (64 x 1024, K=1024)
//   c' = sig(f)*c + sig(i)*tanh(tx*g) ; h' = sig(o)*tanh(c')
// Grid 16 x 256 threads; block owns a 64(M) x 64(N) stripe; 8 wave32s own 2
// of the 16 16x16 tiles each. A slabs arrive via TDM, double-buffered in LDS
// so the tensor DMA for slab s+1 overlaps the WMMA stream for slab s.
__global__ __launch_bounds__(256)
void lstm_cell(const __hip_bfloat16* __restrict__ inp,     // (B,1024) bf16
               const __hip_bfloat16* __restrict__ h_prev,  // (B,H) bf16
               const __hip_bfloat16* __restrict__ Wall,    // (4H,KTOT) bf16
               const __hip_bfloat16* __restrict__ Wt,      // (H,KIN) bf16
               const float* __restrict__ b_all,            // (4H)
               const float* __restrict__ b_t,              // (H)
               float* __restrict__ c_state,                // (B,H) in-place
               float* __restrict__ h_next,                 // (B,H) fp32
               __hip_bfloat16* __restrict__ h_next_bf,     // (B,H) bf16
               float* __restrict__ out)                    // (B,H) or nullptr
{
    __shared__ __hip_bfloat16 sA[2][BB * 256];   // double-buffered 32KB slabs

    const int tid   = threadIdx.x;
    const int lane  = tid & 31;               // wave32
    const int wave  = tid >> 5;               // 0..7
    const int n0    = blockIdx.x * 64;        // N stripe within H
    const int mloc  = lane & 15;
    const int khalf = lane >> 4;
    const unsigned lds_base = (unsigned)(size_t)&sA[0][0]; // low 32b = LDS off

    v8f acc[2][5];
#pragma unroll
    for (int a = 0; a < 2; ++a)
#pragma unroll
        for (int b = 0; b < 5; ++b)
            acc[a][b] = (v8f){0, 0, 0, 0, 0, 0, 0, 0};

    // prologue: kick off slab 0 DMA
    if (wave == 0) tdm_stage_64x256(lds_base, inp);

    for (int s = 0; s < 8; ++s) {
        __syncthreads();   // buffer (s+1)&1 fully consumed (slab s-1's compute)
        if (wave == 0) {
            if (s < 7) {   // issue next slab's DMA, then wait for current only:
                const __hip_bfloat16* src =
                    (s + 1 < 4) ? (inp + (s + 1) * 256)
                                : (h_prev + (s + 1 - 4) * 256);
                tdm_stage_64x256(lds_base + ((s + 1) & 1) * (BB * 256 * 2), src);
                __builtin_amdgcn_s_wait_tensorcnt(1);  // in-order: slab s done
            } else {
                __builtin_amdgcn_s_wait_tensorcnt(0);  // last slab done
            }
        }
        __syncthreads();   // slab s tile visible to all waves

        if (s + 1 < 8) {   // keep next weight slab warm in L2
            const __hip_bfloat16* pf =
                Wall + (size_t)(n0 + (wave & 3) * 16 + mloc) * KTOT + (s + 1) * 256;
            __builtin_prefetch(pf, 0, 1);
        }

        const __hip_bfloat16* buf = sA[s & 1];
        if (s < 4) slab_mma<true >(buf, acc, Wall, Wt, s * 256, n0, wave, mloc, khalf);
        else       slab_mma<false>(buf, acc, Wall, Wt, s * 256, n0, wave, mloc, khalf);
    }

    // ---- fused cell update (C/D layout: M = r + 8*khalf, N = lane&15) ----
#pragma unroll
    for (int ti = 0; ti < 2; ++ti) {
        const int tile = wave + ti * 8;
        const int mt = tile & 3;
        const int nt = tile >> 2;
        const int n  = n0 + nt * 16 + mloc;
        const float bi = b_all[n];
        const float bf = b_all[HH + n];
        const float bo = b_all[2 * HH + n];
        const float bg = b_all[3 * HH + n];
        const float bt = b_t[n];
#pragma unroll
        for (int r = 0; r < 8; ++r) {
            const int m = mt * 16 + r + khalf * 8;
            const float iv = sigmoidf_(acc[ti][0][r] + bi);
            const float fv = sigmoidf_(acc[ti][1][r] + bf);
            const float ov = sigmoidf_(acc[ti][2][r] + bo);
            const float gv = acc[ti][3][r] + bg;
            const float tx = acc[ti][4][r] + bt;
            const float gt = tanhf(tx * gv);
            const size_t idx = (size_t)m * HH + n;
            const float cn = fv * c_state[idx] + iv * gt;
            const float hn = ov * tanhf(cn);
            c_state[idx]   = cn;
            h_next[idx]    = hn;
            h_next_bf[idx] = __float2bfloat16(hn);
            if (out) out[idx] = hn;
        }
    }
}

extern "C" void kernel_launch(void* const* d_in, const int* in_sizes, int n_in,
                              void* d_out, int out_size, void* d_ws, size_t ws_size,
                              hipStream_t stream) {
    const float* x      = (const float*)d_in[0];
    const float* Wall0f = (const float*)d_in[1];
    const float* ball0  = (const float*)d_in[2];
    const float* Wt0f   = (const float*)d_in[3];
    const float* bt0    = (const float*)d_in[4];
    const float* Wall1f = (const float*)d_in[5];
    const float* ball1  = (const float*)d_in[6];
    const float* Wt1f   = (const float*)d_in[7];
    const float* bt1    = (const float*)d_in[8];

    float* out = (float*)d_out;
    const size_t NOUT = (size_t)TSTEPS * BB * HH;     // outputs (T,B,H)
    float* hF = out + NOUT;                           // h_f (L,B,H)
    float* cF = hF + (size_t)2 * BB * HH;             // c_f (L,B,H) -- live c

    // ws layout: bf16 weights (36MB) | bf16 x (64MB) | fp32 h pp (1MB) | bf16 h pp (512KB)
    __hip_bfloat16* wall0 = (__hip_bfloat16*)d_ws;
    __hip_bfloat16* wall1 = wall0 + (size_t)4 * HH * KTOT;
    __hip_bfloat16* wt0   = wall1 + (size_t)4 * HH * KTOT;
    __hip_bfloat16* wt1   = wt0 + (size_t)HH * KIN;
    __hip_bfloat16* xb    = wt1 + (size_t)HH * KIN;           // (T,B,I) bf16
    float* hbuf           = (float*)(xb + (size_t)TSTEPS * BB * II); // [2][L][B][H]
    __hip_bfloat16* hbb   = (__hip_bfloat16*)(hbuf + (size_t)2 * 2 * BB * HH);

    const int nWall4 = (4 * HH * KTOT) / 4;
    const int nWt4   = (HH * KIN) / 4;
    const int nX4    = (TSTEPS * BB * II) / 4;
    cvt_bf16x4_kernel<<<(nWall4 + 255) / 256, 256, 0, stream>>>(Wall0f, wall0, nWall4);
    cvt_bf16x4_kernel<<<(nWall4 + 255) / 256, 256, 0, stream>>>(Wall1f, wall1, nWall4);
    cvt_bf16x4_kernel<<<(nWt4 + 255) / 256, 256, 0, stream>>>(Wt0f, wt0, nWt4);
    cvt_bf16x4_kernel<<<(nWt4 + 255) / 256, 256, 0, stream>>>(Wt1f, wt1, nWt4);
    cvt_bf16x4_kernel<<<(nX4 + 255) / 256, 256, 0, stream>>>(x, xb, nX4);

    hipMemsetAsync(hbuf, 0, (size_t)2 * 2 * BB * HH * sizeof(float), stream);
    hipMemsetAsync(hbb,  0, (size_t)2 * 2 * BB * HH * sizeof(__hip_bfloat16), stream);
    hipMemsetAsync(cF,   0, (size_t)2 * BB * HH * sizeof(float), stream);

    const size_t LBH = (size_t)BB * HH;
    for (int t = 0; t < TSTEPS; ++t) {
        float* hn = hbuf + (size_t)((t + 1) & 1) * 2 * LBH;
        __hip_bfloat16* hcb = hbb + (size_t)(t & 1) * 2 * LBH;
        __hip_bfloat16* hnb = hbb + (size_t)((t + 1) & 1) * 2 * LBH;
        // layer 0: input = x_t (bf16), recurrent = h0(t) (bf16)
        lstm_cell<<<16, 256, 0, stream>>>(
            xb + (size_t)t * BB * II, hcb + 0,
            wall0, wt0, ball0, bt0,
            cF + 0, hn + 0, hnb + 0, (float*)nullptr);
        // layer 1: input = fresh h0(t+1) bf16, recurrent = h1(t) bf16
        lstm_cell<<<16, 256, 0, stream>>>(
            hnb + 0, hcb + LBH,
            wall1, wt1, ball1, bt1,
            cF + LBH, hn + LBH, hnb + LBH, out + (size_t)t * BB * HH);
    }
    // T even -> final fp32 states live in ping buffer 0
    hipMemcpyAsync(hF, hbuf, (size_t)2 * BB * HH * sizeof(float),
                   hipMemcpyDeviceToDevice, stream);
}
